// M3GnetLammps_27771258536734
// MI455X (gfx1250) — matrix operand order repaired and checked by reading, not compiled, not run on hardware
//
#include <hip/hip_runtime.h>
#include <hip/hip_bf16.h>

typedef __bf16 bf16_t;
typedef bf16_t v16bf __attribute__((ext_vector_type(16)));
typedef float  v8f   __attribute__((ext_vector_type(8)));

#define UNITS   128
#define NRBF    10
#define MAXN    3
#define MAXL    3
#define SBF     9
#define NLAYERS 3
#define NTOTAL  60000
#define NLOCAL  50000
#define NGHOST  10000
#define NEDGE   250000
#define NTRIPLE 1000000
#define FEATK   (3*UNITS)          // 384
#define KTILES  (FEATK/32)         // 12
#define NTILES  (UNITS/16)         // 8
#define MTILE   32                 // edges per block (2 WMMA m-tiles)
#define CUTOFF  5.0f
#define TBCUT   4.0f
#define PI_F    3.14159265358979323846f

// ---------------- workspace layout ----------------
constexpr size_t wal(size_t x){ return (x + 255) & ~(size_t)255; }
constexpr size_t SZ_ATTR   = wal((size_t)NTOTAL*UNITS*4);
constexpr size_t SZ_ATTRBF = wal((size_t)NTOTAL*UNITS*2);
constexpr size_t SZ_EATTR  = wal((size_t)NEDGE*UNITS*4);
constexpr size_t SZ_ERBF   = wal((size_t)NEDGE*NRBF*4);
constexpr size_t SZ_ELEN   = wal((size_t)NEDGE*4);
constexpr size_t SZ_EVEC   = wal((size_t)NEDGE*3*4);
constexpr size_t SZ_TBPRE  = wal((size_t)NTRIPLE*SBF*4);
constexpr size_t SZ_TINT   = wal((size_t)NTRIPLE*4);
constexpr size_t SZ_AMO    = wal((size_t)NLOCAL*SBF*4);
constexpr size_t SZ_TBOUT  = wal((size_t)NEDGE*SBF*4);
constexpr int    PERMAT    = NTILES*KTILES*32*16;    // 49152 bf16 elems per matrix
constexpr size_t SZ_WPK    = wal((size_t)12*PERMAT*2);

constexpr size_t OFF_ATTRA  = 0;
constexpr size_t OFF_ATTRB  = OFF_ATTRA  + SZ_ATTR;
constexpr size_t OFF_ATTRBF = OFF_ATTRB  + SZ_ATTR;
constexpr size_t OFF_EATTR  = OFF_ATTRBF + SZ_ATTRBF;
constexpr size_t OFF_ERBF   = OFF_EATTR  + SZ_EATTR;
constexpr size_t OFF_ELEN   = OFF_ERBF   + SZ_ERBF;
constexpr size_t OFF_EVEC   = OFF_ELEN   + SZ_ELEN;
constexpr size_t OFF_TBPRE  = OFF_EVEC   + SZ_EVEC;
constexpr size_t OFF_TCEN   = OFF_TBPRE  + SZ_TBPRE;
constexpr size_t OFF_TIJ    = OFF_TCEN   + SZ_TINT;
constexpr size_t OFF_AMO    = OFF_TIJ    + SZ_TINT;
constexpr size_t OFF_TBOUT  = OFF_AMO    + SZ_AMO;
constexpr size_t OFF_WPK    = OFF_TBOUT  + SZ_TBOUT;

// ---------------- helpers ----------------
__device__ __forceinline__ float sig_(float x){ return 1.0f/(1.0f+__expf(-x)); }
__device__ __forceinline__ float silu_(float x){ return x*sig_(x); }
__device__ __forceinline__ float polycut_(float r, float c){
  float t = r/c, t2 = t*t, t3 = t2*t, t4 = t2*t2, t5 = t4*t;
  float p = 1.0f - 6.0f*t5 + 15.0f*t4 - 10.0f*t3;
  return (r <= c) ? p : 0.0f;
}
__device__ __forceinline__ unsigned int pk_bf16_(float lo, float hi){
  unsigned short ulo = __builtin_bit_cast(unsigned short, (bf16_t)lo);
  unsigned short uhi = __builtin_bit_cast(unsigned short, (bf16_t)hi);
  return (unsigned int)ulo | ((unsigned int)uhi << 16);
}

// ---------------- one-time kernels ----------------
__global__ void k_embed(const int* __restrict__ zn, const float* __restrict__ emb,
                        float* __restrict__ attr){
  int gid = blockIdx.x*blockDim.x + threadIdx.x;
  if (gid >= NTOTAL*UNITS) return;
  int a = gid >> 7, c = gid & 127;
  attr[gid] = emb[zn[a]*UNITS + c];
}

// fp32 attr -> bf16 mirror (halves gather traffic in the WMMA kernel)
__global__ void k_attr_bf(const float* __restrict__ a, unsigned int* __restrict__ o){
  int gid = blockIdx.x*blockDim.x + threadIdx.x;
  if (gid >= NTOTAL*UNITS/2) return;
  float2 f = ((const float2*)a)[gid];
  o[gid] = pk_bf16_(f.x, f.y);
}

__global__ void k_edge_geom(const float* __restrict__ pos, const float* __restrict__ cell,
                            const float* __restrict__ pbc, const int* __restrict__ src,
                            const int* __restrict__ dst, const int* __restrict__ batch,
                            float* __restrict__ evec, float* __restrict__ elen,
                            float* __restrict__ erbf){
  int e = blockIdx.x*blockDim.x + threadIdx.x;
  if (e >= NEDGE) return;
  int s = src[e], d = dst[e];
  const float* C = cell + (size_t)batch[s]*9;
  float o0 = pbc[e*3+0], o1 = pbc[e*3+1], o2 = pbc[e*3+2];
  float vx = pos[s*3+0] - (pos[d*3+0] + o0*C[0] + o1*C[3] + o2*C[6]);
  float vy = pos[s*3+1] - (pos[d*3+1] + o0*C[1] + o1*C[4] + o2*C[7]);
  float vz = pos[s*3+2] - (pos[d*3+2] + o0*C[2] + o1*C[5] + o2*C[8]);
  float r = sqrtf(vx*vx + vy*vy + vz*vz);
  evec[e*3+0] = vx; evec[e*3+1] = vy; evec[e*3+2] = vz; elen[e] = r;
  #pragma unroll
  for (int k = 0; k < NRBF; k++){
    float mu = (float)k * (CUTOFF/(float)(NRBF-1));
    float dm = r - mu;
    erbf[e*NRBF+k] = __expf(-2.0f*dm*dm);           // exp(-(r-mu)^2 / 0.5)
  }
}

__global__ void k_edge_enc(const float* __restrict__ erbf, const float* __restrict__ w,
                           const float* __restrict__ b, float* __restrict__ eattr){
  int gid = blockIdx.x*blockDim.x + threadIdx.x;
  if (gid >= NEDGE*UNITS) return;
  int e = gid >> 7, c = gid & 127;
  float acc = b[c];
  #pragma unroll
  for (int k = 0; k < NRBF; k++) acc += erbf[e*NRBF+k] * w[k*UNITS+c];
  eattr[gid] = acc;
}

__global__ void k_triple(const int* __restrict__ tbi, const int* __restrict__ src,
                         const float* __restrict__ evec, const float* __restrict__ elen,
                         float* __restrict__ tbpre, int* __restrict__ tcen,
                         int* __restrict__ tij){
  int t = blockIdx.x*blockDim.x + threadIdx.x;
  if (t >= NTRIPLE) return;
  int ij = tbi[2*t], ik = tbi[2*t+1];
  float rij = elen[ij], rik = elen[ik];
  float dv = evec[ij*3+0]*evec[ik*3+0] + evec[ij*3+1]*evec[ik*3+1] + evec[ij*3+2]*evec[ik*3+2];
  float cosv = dv / (rij*rik + 1e-12f);
  cosv = fminf(fmaxf(cosv, -1.0f + 1e-7f), 1.0f - 1e-7f);
  float th = acosf(cosv);
  float cut = polycut_(rij, TBCUT) * polycut_(rik, TBCUT);
  float inv = 1.0f / (rik + 1e-6f);
  #pragma unroll
  for (int n = 0; n < MAXN; n++){
    float rad = __sinf((float)(n+1)*PI_F*rik/TBCUT) * inv;
    #pragma unroll
    for (int l = 0; l < MAXL; l++){
      tbpre[t*SBF + n*MAXL + l] = rad * __cosf((float)l*th) * cut;
    }
  }
  tcen[t] = src[ik];
  tij[t]  = ij;
}

// repack 384x128 fp32 weights into bf16 WMMA B-fragments:
// elem e of lane ln in (ntile,ktile) holds W[ktile*32 + (ln>>4)*16 + e][ntile*16 + (ln&15)]
__global__ void k_pack(const float* __restrict__ gewm, const float* __restrict__ gewg,
                       const float* __restrict__ gawm, const float* __restrict__ gawg,
                       bf16_t* __restrict__ wpk){
  int gid = blockIdx.x*blockDim.x + threadIdx.x;
  if (gid >= 12*PERMAT) return;
  int slot = gid / PERMAT;
  int r    = gid - slot*PERMAT;
  int e    = r & 15;
  int ln   = (r >> 4) & 31;
  int ntk  = r >> 9;                 // ntile*KTILES + ktile
  int kt   = ntk % KTILES;
  int nt   = ntk / KTILES;
  int L = slot >> 2, stage = (slot >> 1) & 1, mat = slot & 1;
  const float* W = (stage == 0) ? (mat == 0 ? gewm : gewg)
                                : (mat == 0 ? gawm : gawg);
  int k = kt*32 + (ln >> 4)*16 + e;
  int n = nt*16 + (ln & 15);
  wpk[gid] = (bf16_t)W[(size_t)L*FEATK*UNITS + (size_t)k*UNITS + n];
}

// ---------------- per-layer kernels ----------------
__global__ void k_tb_atom(const float* __restrict__ attr, const float* __restrict__ w,
                          const float* __restrict__ b, float* __restrict__ amo){
  int gid = blockIdx.x*blockDim.x + threadIdx.x;
  if (gid >= NLOCAL*SBF) return;
  int a = gid / SBF, s = gid - a*SBF;
  float acc = b[s];
  #pragma unroll 8
  for (int k = 0; k < UNITS; k++) acc += attr[a*UNITS+k] * w[k*SBF+s];
  amo[gid] = sig_(acc);
}

__global__ void k_tb_scatter(const float* __restrict__ tbpre, const int* __restrict__ tcen,
                             const int* __restrict__ tij, const float* __restrict__ amo,
                             float* __restrict__ tbout){
  int t = blockIdx.x*blockDim.x + threadIdx.x;
  if (t >= NTRIPLE) return;
  int c = tcen[t], m = tij[t];
  #pragma unroll
  for (int s = 0; s < SBF; s++)
    atomicAdd(&tbout[m*SBF+s], tbpre[t*SBF+s] * amo[c*SBF+s]);
}

__global__ void k_edge_tb(const float* __restrict__ tbout, const float* __restrict__ w,
                          const float* __restrict__ b, float* __restrict__ eattr){
  int gid = blockIdx.x*blockDim.x + threadIdx.x;
  if (gid >= NEDGE*UNITS) return;
  int e = gid >> 7, c = gid & 127;
  float acc = b[c];
  #pragma unroll
  for (int s = 0; s < SBF; s++) acc += tbout[e*SBF+s] * w[s*UNITS+c];
  eattr[gid] += acc;
}

// fused gated MLP, 32-edge tile / block, 8 waves, one N-tile per wave, 2 m-tiles each.
// A fragments pre-packed to bf16 in LDS cooperatively (48B lane stride: 16B aligned,
// 12-bank skew); B fragments streamed from prepacked weights (L2-resident).
#define APITCH 12   // u32 per (ktile,mtile,lane) slot: 8 data + 4 pad
__global__ __launch_bounds__(256) void k_gated_mlp(
    const bf16_t* __restrict__ attr_bf, float* __restrict__ edge_attr,
    float* __restrict__ atom_nxt,
    const int* __restrict__ src, const int* __restrict__ dst,
    const float* __restrict__ erbf,
    const bf16_t* __restrict__ wm_pk, const bf16_t* __restrict__ wg_pk,
    const float* __restrict__ bm, const float* __restrict__ bg,
    const float* __restrict__ elw, const float* __restrict__ elb,
    int stage)
{
  __shared__ unsigned int s_apack[KTILES*2*32*APITCH];  // 36864 B
  __shared__ float s_el[MTILE*UNITS];                   // 16384 B
  __shared__ int   s_src[MTILE];
  __shared__ int   s_dst[MTILE];
  const int tid   = threadIdx.x;
  const int lane  = tid & 31;
  const int wv    = tid >> 5;
  const int ebase = blockIdx.x * MTILE;

  if (tid < MTILE){
    int e = min(ebase + tid, NEDGE - 1);
    s_src[tid] = src[e];
    s_dst[tid] = dst[e];
  }
  __syncthreads();

  // cooperative A-fragment pack: 12kt x 2mt x 32ln x 8 u32-pairs = 6144 stores
  for (int i = tid; i < KTILES*2*32*8; i += 256){
    int p    = i & 7;
    int ln   = (i >> 3) & 31;
    int mt   = (i >> 8) & 1;
    int kt   = i >> 9;
    int slot = (kt*2 + mt)*32 + ln;
    int e0   = 2*p;
    int k    = kt*32 + (ln >> 4)*16 + e0;
    int m    = mt*16 + (ln & 15);
    unsigned int bits;
    if (k < 2*UNITS){
      int row = (k < UNITS) ? s_src[m] : s_dst[m];
      int col = k & 127;
      bits = *(const unsigned int*)(attr_bf + (size_t)row*UNITS + col); // bf16 pair copy
    } else {
      int e = min(ebase + m, NEDGE - 1);
      float2 f = *(const float2*)(edge_attr + (size_t)e*UNITS + (k - 2*UNITS));
      bits = pk_bf16_(f.x, f.y);
    }
    s_apack[slot*APITCH + p] = bits;
  }
  // rbf-conditioned elementwise projection: elproj[m][n] = elb[n] + rbf[e]·elw[:,n]
  for (int i = tid; i < MTILE*UNITS; i += 256){
    int m = i >> 7, n = i & 127;
    int e = min(ebase + m, NEDGE - 1);
    float acc = elb[n];
    #pragma unroll
    for (int k = 0; k < NRBF; k++) acc += erbf[e*NRBF+k] * elw[k*UNITS+n];
    s_el[i] = acc;
  }
  __syncthreads();

  const int ntile = wv;
  v8f am0 = {}, ag0 = {}, am1 = {}, ag1 = {};

  __builtin_prefetch(wm_pk + (size_t)(ntile*KTILES)*32*16, 0, 1);
  __builtin_prefetch(wg_pk + (size_t)(ntile*KTILES)*32*16, 0, 1);

  #pragma unroll
  for (int kt = 0; kt < KTILES; kt++){
    v16bf a0 = *(const v16bf*)&s_apack[((kt*2 + 0)*32 + lane)*APITCH];
    v16bf a1 = *(const v16bf*)&s_apack[((kt*2 + 1)*32 + lane)*APITCH];
    const int bidx = (ntile*KTILES + kt)*32 + lane;
    v16bf bmv = ((const v16bf*)wm_pk)[bidx];
    v16bf bgv = ((const v16bf*)wg_pk)[bidx];
    am0 = __builtin_amdgcn_wmma_f32_16x16x32_bf16(false, a0, false, bmv, (short)0, am0, false, false);
    ag0 = __builtin_amdgcn_wmma_f32_16x16x32_bf16(false, a0, false, bgv, (short)0, ag0, false, false);
    am1 = __builtin_amdgcn_wmma_f32_16x16x32_bf16(false, a1, false, bmv, (short)0, am1, false, false);
    ag1 = __builtin_amdgcn_wmma_f32_16x16x32_bf16(false, a1, false, bgv, (short)0, ag1, false, false);
  }

  // epilogue: D layout -> m = v + (lane>>4)*8 (+16 for tile1), n = ntile*16 + (lane&15)
  const int ncol  = ntile*16 + (lane & 15);
  const float bim = bm[ncol];
  const float big = bg[ncol];
  const int mhalf = (lane >> 4) * 8;
  #pragma unroll
  for (int v = 0; v < 8; v++){
    #pragma unroll
    for (int half = 0; half < 2; half++){
      int   m  = half*16 + mhalf + v;
      float xm = (half ? am1[v] : am0[v]) + bim;
      float xg = (half ? ag1[v] : ag0[v]) + big;
      int   e  = ebase + m;
      if (e < NEDGE){
        float out = silu_(xm) * sig_(xg) * s_el[m*UNITS + ncol];
        if (stage == 0){
          float* p = &edge_attr[(size_t)e*UNITS + ncol];
          *p = *p + out;
        } else {
          atomicAdd(&atom_nxt[(size_t)s_src[m]*UNITS + ncol], out);
        }
      }
    }
  }
}

__global__ void k_ghost(const int* __restrict__ gmap, float* __restrict__ attr){
  int gid = blockIdx.x*blockDim.x + threadIdx.x;
  if (gid >= NGHOST*UNITS) return;
  int g = gid >> 7, c = gid & 127;
  attr[(size_t)(NLOCAL + g)*UNITS + c] = attr[(size_t)gmap[g]*UNITS + c];
}

__global__ __launch_bounds__(128) void k_final(
    const float* __restrict__ attr, const float* __restrict__ w1,
    const float* __restrict__ b1, const float* __restrict__ w2,
    const float* __restrict__ b2, const float* __restrict__ esc,
    const float* __restrict__ esh, const int* __restrict__ zn,
    float* __restrict__ out){
  __shared__ float row[UNITS];
  __shared__ float red[UNITS];
  int a = blockIdx.x, c = threadIdx.x;
  row[c] = attr[(size_t)a*UNITS + c];
  __syncthreads();
  float acc = b1[c];
  #pragma unroll 8
  for (int k = 0; k < UNITS; k++) acc += row[k] * w1[k*UNITS + c];
  red[c] = silu_(acc) * w2[c];
  __syncthreads();
  for (int s = 64; s > 0; s >>= 1){
    if (c < s) red[c] += red[c + s];
    __syncthreads();
  }
  if (c == 0){
    int z = zn[a];
    atomicAdd(out, (red[0] + b2[0]) * esc[z] + esh[z]);
  }
}

// ---------------- launch ----------------
extern "C" void kernel_launch(void* const* d_in, const int* in_sizes, int n_in,
                              void* d_out, int out_size, void* d_ws, size_t ws_size,
                              hipStream_t stream) {
  (void)in_sizes; (void)n_in; (void)out_size; (void)ws_size;
  const int*   zn    = (const int*)  d_in[0];
  const float* pos   = (const float*)d_in[1];
  const float* cell  = (const float*)d_in[2];
  const float* pbc   = (const float*)d_in[3];
  const int*   esrc  = (const int*)  d_in[4];
  const int*   edst  = (const int*)  d_in[5];
  const int*   tbi   = (const int*)  d_in[6];
  const int*   batch = (const int*)  d_in[7];
  const int*   gmap  = (const int*)  d_in[8];
  /* d_in[9] = nlocal scalar (compile-time constant NLOCAL) */
  const float* emb   = (const float*)d_in[10];
  const float* encw  = (const float*)d_in[11];
  const float* encb  = (const float*)d_in[12];
  const float* tbaw  = (const float*)d_in[13];
  const float* tbab  = (const float*)d_in[14];
  const float* tbgw  = (const float*)d_in[15];
  const float* tbgb  = (const float*)d_in[16];
  const float* gewm  = (const float*)d_in[17];
  const float* gebm  = (const float*)d_in[18];
  const float* gewg  = (const float*)d_in[19];
  const float* gebg  = (const float*)d_in[20];
  const float* elew  = (const float*)d_in[21];
  const float* eleb  = (const float*)d_in[22];
  const float* gawm  = (const float*)d_in[23];
  const float* gabm  = (const float*)d_in[24];
  const float* gawg  = (const float*)d_in[25];
  const float* gabg  = (const float*)d_in[26];
  const float* elaw  = (const float*)d_in[27];
  const float* elab  = (const float*)d_in[28];
  const float* fw1   = (const float*)d_in[29];
  const float* fb1   = (const float*)d_in[30];
  const float* fw2   = (const float*)d_in[31];
  const float* fb2   = (const float*)d_in[32];
  const float* escl  = (const float*)d_in[33];
  const float* esft  = (const float*)d_in[34];

  char* ws = (char*)d_ws;
  float*        attrA  = (float*)       (ws + OFF_ATTRA);
  float*        attrB  = (float*)       (ws + OFF_ATTRB);
  unsigned int* attrBF = (unsigned int*)(ws + OFF_ATTRBF);
  float*        eattr  = (float*)       (ws + OFF_EATTR);
  float*        erbf   = (float*)       (ws + OFF_ERBF);
  float*        elen   = (float*)       (ws + OFF_ELEN);
  float*        evec   = (float*)       (ws + OFF_EVEC);
  float*        tbpre  = (float*)       (ws + OFF_TBPRE);
  int*          tcen   = (int*)         (ws + OFF_TCEN);
  int*          tij    = (int*)         (ws + OFF_TIJ);
  float*        amo    = (float*)       (ws + OFF_AMO);
  float*        tbout  = (float*)       (ws + OFF_TBOUT);
  bf16_t*       wpk    = (bf16_t*)      (ws + OFF_WPK);

  const int B = 256;
  auto cdiv = [](int a, int b){ return (a + b - 1) / b; };

  k_pack<<<cdiv(12*PERMAT, B), B, 0, stream>>>(gewm, gewg, gawm, gawg, wpk);
  k_embed<<<cdiv(NTOTAL*UNITS, B), B, 0, stream>>>(zn, emb, attrA);
  k_edge_geom<<<cdiv(NEDGE, B), B, 0, stream>>>(pos, cell, pbc, esrc, edst, batch,
                                                evec, elen, erbf);
  k_edge_enc<<<cdiv(NEDGE*UNITS, B), B, 0, stream>>>(erbf, encw, encb, eattr);
  k_triple<<<cdiv(NTRIPLE, B), B, 0, stream>>>(tbi, esrc, evec, elen, tbpre, tcen, tij);

  for (int L = 0; L < NLAYERS; L++){
    float* cur = (L & 1) ? attrB : attrA;
    float* nxt = (L & 1) ? attrA : attrB;

    // bf16 mirror of current atom features (used by both gated-MLP stages)
    k_attr_bf<<<cdiv(NTOTAL*UNITS/2, B), B, 0, stream>>>(cur, attrBF);

    k_tb_atom<<<cdiv(NLOCAL*SBF, B), B, 0, stream>>>(cur, tbaw + (size_t)L*UNITS*SBF,
                                                     tbab + (size_t)L*SBF, amo);
    hipMemsetAsync(tbout, 0, (size_t)NEDGE*SBF*sizeof(float), stream);
    k_tb_scatter<<<cdiv(NTRIPLE, B), B, 0, stream>>>(tbpre, tcen, tij, amo, tbout);
    k_edge_tb<<<cdiv(NEDGE*UNITS, B), B, 0, stream>>>(tbout, tbgw + (size_t)L*SBF*UNITS,
                                                      tbgb + (size_t)L*UNITS, eattr);
    // edge gated MLP (stage 0): edge_attr += silu(m)*sig(g) * elproj
    k_gated_mlp<<<cdiv(NEDGE, MTILE), 256, 0, stream>>>(
        (const bf16_t*)attrBF, eattr, nullptr, esrc, edst, erbf,
        wpk + (size_t)(L*4 + 0)*PERMAT, wpk + (size_t)(L*4 + 1)*PERMAT,
        gebm + (size_t)L*UNITS, gebg + (size_t)L*UNITS,
        elew + (size_t)L*NRBF*UNITS, eleb + (size_t)L*UNITS, 0);
    // atom_next = atom_attr[:nlocal], then scatter-add the atom gated MLP
    hipMemcpyAsync(nxt, cur, (size_t)NLOCAL*UNITS*sizeof(float),
                   hipMemcpyDeviceToDevice, stream);
    k_gated_mlp<<<cdiv(NEDGE, MTILE), 256, 0, stream>>>(
        (const bf16_t*)attrBF, eattr, nxt, esrc, edst, erbf,
        wpk + (size_t)(L*4 + 2)*PERMAT, wpk + (size_t)(L*4 + 3)*PERMAT,
        gabm + (size_t)L*UNITS, gabg + (size_t)L*UNITS,
        elaw + (size_t)L*NRBF*UNITS, elab + (size_t)L*UNITS, 1);
    if (L < NLAYERS - 1)
      k_ghost<<<cdiv(NGHOST*UNITS, B), B, 0, stream>>>(gmap, nxt);
  }

  hipMemsetAsync(d_out, 0, sizeof(float), stream);
  // after 3 layers the live buffer is attrB (L2: cur=attrA -> nxt=attrB)
  k_final<<<NLOCAL, 128, 0, stream>>>(attrB, fw1, fb1, fw2, fb2, escl, esft, zn,
                                      (float*)d_out);
}
